// GraphModel_1108101562622
// MI455X (gfx1250) — matrix operand order repaired and checked by reading, not compiled, not run on hardware
//
#include <hip/hip_runtime.h>
#include <hip/hip_bf16.h>
#include <stdint.h>

// ---------------- problem constants (from reference) ----------------
#define NATTRS 115
#define NOBJS  245
#define NPAIRS 28175
#define NNODES (NATTRS + NOBJS + NPAIRS)   // 28535
#define FEAT   512
#define HID    2048
#define OUTF   512
#define NEDGES (4 * NNODES)                // 114140
#define BATCH  512

typedef __attribute__((ext_vector_type(16))) __bf16 v16bf;
typedef __attribute__((ext_vector_type(8)))  __bf16 v8bf;
typedef __attribute__((ext_vector_type(8)))  float  v8f;
typedef __attribute__((ext_vector_type(4)))  unsigned int u32x4;
typedef __attribute__((ext_vector_type(8)))  int          i32x8;
typedef __attribute__((ext_vector_type(4)))  int          i32x4;

// ---------------- small elementwise kernels ----------------
__global__ void k_fill_f32(float* p, int n, float v) {
    int i = blockIdx.x * blockDim.x + threadIdx.x;
    if (i < n) p[i] = v;
}

__global__ void k_deg_atomic(const long long* __restrict__ dst, float* __restrict__ deg, int ne) {
    int e = blockIdx.x * blockDim.x + threadIdx.x;
    if (e < ne) atomicAdd(&deg[(int)dst[e]], 1.0f);
}

__global__ void k_dinv(const float* __restrict__ deg, float* __restrict__ dinv, int n) {
    int i = blockIdx.x * blockDim.x + threadIdx.x;
    if (i < n) dinv[i] = __frsqrt_rn(deg[i]);   // deg >= 1 always (self loops)
}

__global__ void k_edge_norm(const long long* __restrict__ src, const long long* __restrict__ dst,
                            const float* __restrict__ dinv, float* __restrict__ norm, int ne) {
    int e = blockIdx.x * blockDim.x + threadIdx.x;
    if (e < ne) norm[e] = dinv[(int)src[e]] * dinv[(int)dst[e]];
}

__global__ void k_f32_to_bf16(const float* __restrict__ in, __bf16* __restrict__ out, int n) {
    int i = blockIdx.x * blockDim.x + threadIdx.x;
    if (i < n) out[i] = (__bf16)in[i];
}

// agg[i,f] = h[i,f] * dinv[i]^2   (self-loop contribution, also initializes agg)
__global__ void k_agg_selfinit(const float* __restrict__ h, const float* __restrict__ dinv,
                               float* __restrict__ agg, int n, int f) {
    int idx = blockIdx.x * blockDim.x + threadIdx.x;
    if (idx < n * f) {
        int row = idx / f;
        float s = dinv[row];
        agg[idx] = h[idx] * (s * s);
    }
}

// one block per edge: agg[dst] += h[src] * norm[e]
__global__ __launch_bounds__(256) void k_agg_edges(const float* __restrict__ h,
                                                   const long long* __restrict__ src,
                                                   const long long* __restrict__ dst,
                                                   const float* __restrict__ norm,
                                                   float* __restrict__ agg, int f) {
    int e = blockIdx.x;
    int s = (int)src[e];
    int d = (int)dst[e];
    float w = norm[e];
    const float* hs = h + (size_t)s * f;
    float* ad = agg + (size_t)d * f;
    for (int c = threadIdx.x; c < f; c += blockDim.x)
        atomicAdd(&ad[c], hs[c] * w);
}

// x_next(bf16) = relu(agg + b)
__global__ void k_bias_relu_bf16(const float* __restrict__ agg, const float* __restrict__ b,
                                 __bf16* __restrict__ out, int n, int f) {
    int idx = blockIdx.x * blockDim.x + threadIdx.x;
    if (idx < n * f) {
        float v = agg[idx] + b[idx % f];
        out[idx] = (__bf16)(v > 0.0f ? v : 0.0f);
    }
}

// emb = agg + b (in place, no relu)
__global__ void k_bias_f32(float* __restrict__ agg, const float* __restrict__ b, int n, int f) {
    int idx = blockIdx.x * blockDim.x + threadIdx.x;
    if (idx < n * f) agg[idx] += b[idx % f];
}

// L2-normalize one pair row of emb, write transposed bf16: embT[k*NPAIRS + j]
__global__ __launch_bounds__(256) void k_l2norm_pairs_T(const float* __restrict__ emb,
                                                        __bf16* __restrict__ embT) {
    __shared__ float red[256];
    int j = blockIdx.x;                       // pair index 0..NPAIRS-1
    const float* row = emb + (size_t)(NATTRS + NOBJS + j) * OUTF;
    int t = threadIdx.x;
    float a = row[t], c = row[t + 256];
    red[t] = a * a + c * c;
    __syncthreads();
    for (int s = 128; s > 0; s >>= 1) {
        if (t < s) red[t] += red[t + s];
        __syncthreads();
    }
    float scale = 1.0f / fmaxf(sqrtf(red[0]), 1e-12f);
    embT[(size_t)t * NPAIRS + j]         = (__bf16)(a * scale);
    embT[(size_t)(t + 256) * NPAIRS + j] = (__bf16)(c * scale);
}

// L2-normalize img rows -> bf16 row-major
__global__ __launch_bounds__(256) void k_l2norm_img(const float* __restrict__ img,
                                                    __bf16* __restrict__ imgb) {
    __shared__ float red[256];
    int i = blockIdx.x;
    const float* row = img + (size_t)i * OUTF;
    int t = threadIdx.x;
    float a = row[t], c = row[t + 256];
    red[t] = a * a + c * c;
    __syncthreads();
    for (int s = 128; s > 0; s >>= 1) {
        if (t < s) red[t] += red[t + s];
        __syncthreads();
    }
    float scale = 1.0f / fmaxf(sqrtf(red[0]), 1e-12f);
    imgb[(size_t)i * OUTF + t]       = (__bf16)(a * scale);
    imgb[(size_t)i * OUTF + t + 256] = (__bf16)(c * scale);
}

// ---------------- TDM: 2D tensor tile -> LDS (zero-fills OOB) ----------------
#if __has_builtin(__builtin_amdgcn_tensor_load_to_lds)
#define USE_TDM 1
// rem0/rem1: remaining tensor extent (elements) from tile origin in dim0/dim1
// tile0/tile1: tile shape (elements); stride0: row stride in elements (2-byte data)
__device__ __forceinline__ void tdm_load_2d_bf16(unsigned int lds_off, const void* gaddr,
                                                 unsigned int rem0, unsigned int rem1,
                                                 unsigned int tile0, unsigned int tile1,
                                                 unsigned int stride0) {
    unsigned long long ga = (unsigned long long)gaddr;
    u32x4 g0;
    g0[0] = 1u;                                             // count=1 (valid), user mode
    g0[1] = lds_off;                                        // lds_addr (bytes)
    g0[2] = (unsigned int)(ga & 0xFFFFFFFFu);               // global_addr[31:0]
    g0[3] = (unsigned int)((ga >> 32) & 0x1FFFFFFu) | (2u << 30);  // addr[56:32], type=2
    i32x8 g1;
    g1[0] = (int)(1u << 16);                                // data_size=1 (2 bytes), mask=0
    g1[1] = (int)((rem0 & 0xFFFFu) << 16);                  // tensor_dim0[15:0] @bit48
    g1[2] = (int)(((rem0 >> 16) & 0xFFFFu) | ((rem1 & 0xFFFFu) << 16));   // dim0 hi | dim1 lo
    g1[3] = (int)(((rem1 >> 16) & 0xFFFFu) | ((tile0 & 0xFFFFu) << 16));  // dim1 hi | tile_dim0
    g1[4] = (int)(tile1 & 0xFFFFu);                         // tile_dim1 (tile_dim2=0)
    g1[5] = (int)stride0;                                   // tensor_dim0_stride[31:0]
    g1[6] = 0;                                              // stride hi / dim1_stride (2D: unused)
    g1[7] = 0;
    i32x4 z4 = {0, 0, 0, 0};
    i32x8 z8 = {0, 0, 0, 0, 0, 0, 0, 0};
    // 6-arg toolchain variant: (g0, g1, g2, g3, g4, cpol)
    __builtin_amdgcn_tensor_load_to_lds(g0, g1, z4, z4, z8, 0);
}
#else
#define USE_TDM 0
#endif

// ---------------- WMMA bf16 GEMM: C(MxN,f32) = A(MxK,bf16) @ B(KxN,bf16) ----------------
// 256 threads = 8 waves (4x2). Block tile 128x64, K-step 32. Wave tile 32x32 (2x2 WMMA).
// TDM path: double-buffered LDS; DMA of tile i+1 overlaps WMMA compute of tile i.
#define BM 128
#define BN 64
#define BK 32

__global__ __launch_bounds__(256) void k_gemm_bf16(const __bf16* __restrict__ A,
                                                   const __bf16* __restrict__ B,
                                                   float* __restrict__ C,
                                                   int M, int N, int K) {
    __shared__ __bf16 As[2][BM * BK];   // 2 x 8 KB
    __shared__ __bf16 Bs[2][BK * BN];   // 2 x 4 KB

    const int m0 = blockIdx.x * BM;
    const int n0 = blockIdx.y * BN;
    const int t = threadIdx.x;
    const int wave = t >> 5;
    const int lane = t & 31;
    const int wm = wave >> 1;        // 0..3  -> M offset wm*32
    const int wn = wave & 1;         // 0..1  -> N offset wn*32
    const int lhalf = lane >> 4;     // 0/1
    const int lrow = lane & 15;

    v8f acc[2][2] = {};

#if USE_TDM
    unsigned int as_off[2], bs_off[2];
    as_off[0] = (unsigned int)(uintptr_t)(void*)As[0];   // low 32b = LDS offset
    as_off[1] = (unsigned int)(uintptr_t)(void*)As[1];
    bs_off[0] = (unsigned int)(uintptr_t)(void*)Bs[0];
    bs_off[1] = (unsigned int)(uintptr_t)(void*)Bs[1];

    // prologue: DMA tile 0 into buffer 0 (wave 0 only; TDM zero-fills OOB)
    if (t < 32) {
        tdm_load_2d_bf16(as_off[0], A + (size_t)m0 * K,
                         (unsigned int)K, (unsigned int)(M - m0), BK, BM, (unsigned int)K);
        tdm_load_2d_bf16(bs_off[0], B + n0,
                         (unsigned int)(N - n0), (unsigned int)K, BN, BK, (unsigned int)N);
    }
    int cur = 0;
#endif

    for (int k0 = 0; k0 < K; k0 += BK) {
#if USE_TDM
        if (t < 32)
            __builtin_amdgcn_s_wait_tensorcnt(0);   // buffer `cur` filled
        // single barrier: publishes buffer `cur` AND guarantees every wave has
        // finished reading buffer `cur^1` (iteration i-1) before we overwrite it
        __syncthreads();
        int k1 = k0 + BK;
        if (k1 < K && t < 32) {
            int nxt = cur ^ 1;
            tdm_load_2d_bf16(as_off[nxt], A + (size_t)m0 * K + k1,
                             (unsigned int)(K - k1), (unsigned int)(M - m0),
                             BK, BM, (unsigned int)K);
            tdm_load_2d_bf16(bs_off[nxt], B + (size_t)k1 * N + n0,
                             (unsigned int)(N - n0), (unsigned int)(K - k1),
                             BN, BK, (unsigned int)N);
        }
        const __bf16* Asc = As[cur];
        const __bf16* Bsc = Bs[cur];
#else
        // ---- manual staging fallback (single buffer, two barriers) ----
        const __bf16* Asc = As[0];
        const __bf16* Bsc = Bs[0];
        #pragma unroll
        for (int c = 0; c < 2; ++c) {
            int chunk = t + c * 256;          // 0..511
            int row = chunk >> 2;             // 0..127
            int kc = (chunk & 3) << 3;        // 0,8,16,24
            int gr = m0 + row;
            uint4 v = make_uint4(0u, 0u, 0u, 0u);
            if (gr < M) v = *(const uint4*)(A + (size_t)gr * K + k0 + kc);
            *(uint4*)(As[0] + row * BK + kc) = v;
        }
        {
            int row = t >> 3;                 // k 0..31
            int nc = (t & 7) << 3;            // 0..56
            int gn = n0 + nc;
            const __bf16* bp = B + (size_t)(k0 + row) * N + gn;
            uint4 v = make_uint4(0u, 0u, 0u, 0u);
            if (gn + 8 <= N) {
                v = *(const uint4*)bp;
            } else if (gn < N) {
                __bf16 tmp[8];
                #pragma unroll
                for (int i = 0; i < 8; ++i) tmp[i] = (gn + i < N) ? bp[i] : (__bf16)0.0f;
                v = *(uint4*)tmp;
            }
            *(uint4*)(Bs[0] + row * BN + nc) = v;
        }
        __syncthreads();
#endif

        // ---- load fragments per ISA layouts ----
        // A 16x32 bf16: lane<16 holds M=lrow, K = {0..7, 16..23}; lane>=16: K = {8..15, 24..31}
        v16bf af[2];
        #pragma unroll
        for (int i = 0; i < 2; ++i) {
            const __bf16* ap = Asc + (wm * 32 + i * 16 + lrow) * BK;
            v8bf lo = *(const v8bf*)(ap + 8 * lhalf);
            v8bf hi = *(const v8bf*)(ap + 16 + 8 * lhalf);
            af[i] = __builtin_shufflevector(lo, hi, 0, 1, 2, 3, 4, 5, 6, 7,
                                            8, 9, 10, 11, 12, 13, 14, 15);
        }
        // B 32x16 bf16: lane holds N=lrow, K = {0..7, 16..23} (lane<16) / {8..15, 24..31} (lane>=16)
        v16bf bfr[2];
        #pragma unroll
        for (int j = 0; j < 2; ++j)
            #pragma unroll
            for (int s = 0; s < 16; ++s)
                bfr[j][s] = Bsc[((s < 8) ? (8 * lhalf + s) : (16 + 8 * lhalf + (s - 8))) * BN + wn * 32 + j * 16 + lrow];

        #pragma unroll
        for (int i = 0; i < 2; ++i)
            #pragma unroll
            for (int j = 0; j < 2; ++j)
                acc[i][j] = __builtin_amdgcn_wmma_f32_16x16x32_bf16(
                    false, af[i], false, bfr[j], (short)0, acc[i][j], false, false);

#if USE_TDM
        cur ^= 1;
#else
        __syncthreads();
#endif
    }

    // ---- store C: VGPR r -> M = r + 8*lhalf, N = lrow (per 16x16 f32 layout) ----
    #pragma unroll
    for (int i = 0; i < 2; ++i) {
        #pragma unroll
        for (int j = 0; j < 2; ++j) {
            int col = n0 + wn * 32 + j * 16 + lrow;
            if (col < N) {
                int rbase = m0 + wm * 32 + i * 16 + 8 * lhalf;
                #pragma unroll
                for (int r = 0; r < 8; ++r) {
                    int row = rbase + r;
                    if (row < M) C[(size_t)row * N + col] = acc[i][j][r];
                }
            }
        }
    }
}

// ---------------- host-side orchestration ----------------
static inline size_t align256(size_t x) { return (x + 255) & ~(size_t)255; }

extern "C" void kernel_launch(void* const* d_in, const int* in_sizes, int n_in,
                              void* d_out, int out_size, void* d_ws, size_t ws_size,
                              hipStream_t stream) {
    (void)in_sizes; (void)n_in; (void)out_size; (void)ws_size;

    const float*     img   = (const float*)d_in[0];
    const float*     nodes = (const float*)d_in[1];
    const long long* ei    = (const long long*)d_in[2];   // int64 [2, NEDGES]
    const float*     W1    = (const float*)d_in[3];
    const float*     b1    = (const float*)d_in[4];
    const float*     W2    = (const float*)d_in[5];
    const float*     b2    = (const float*)d_in[6];
    const float*     W3    = (const float*)d_in[7];
    const float*     b3    = (const float*)d_in[8];
    float*           out   = (float*)d_out;

    const long long* src = ei;
    const long long* dst = ei + NEDGES;

    // carve workspace
    char* ws = (char*)d_ws;
    size_t off = 0;
    auto carve = [&](size_t bytes) { char* p = ws + off; off += align256(bytes); return p; };
    float*  deg   = (float*)carve(sizeof(float) * NNODES);
    float*  dinv  = (float*)carve(sizeof(float) * NNODES);
    float*  norm  = (float*)carve(sizeof(float) * NEDGES);
    __bf16* W1b   = (__bf16*)carve(sizeof(__bf16) * FEAT * HID);
    __bf16* W2b   = (__bf16*)carve(sizeof(__bf16) * HID * HID);
    __bf16* W3b   = (__bf16*)carve(sizeof(__bf16) * HID * OUTF);
    __bf16* xb    = (__bf16*)carve(sizeof(__bf16) * (size_t)NNODES * HID);
    float*  h     = (float*)carve(sizeof(float) * (size_t)NNODES * HID);
    float*  agg   = (float*)carve(sizeof(float) * (size_t)NNODES * HID);
    __bf16* embT  = (__bf16*)carve(sizeof(__bf16) * (size_t)OUTF * NPAIRS);
    __bf16* imgb  = (__bf16*)carve(sizeof(__bf16) * (size_t)BATCH * OUTF);

    const int T = 256;
    auto cdiv = [](long long a, long long b) { return (int)((a + b - 1) / b); };

    // ---- graph normalization ----
    k_fill_f32<<<cdiv(NNODES, T), T, 0, stream>>>(deg, NNODES, 1.0f);       // self loops
    k_deg_atomic<<<cdiv(NEDGES, T), T, 0, stream>>>(dst, deg, NEDGES);
    k_dinv<<<cdiv(NNODES, T), T, 0, stream>>>(deg, dinv, NNODES);
    k_edge_norm<<<cdiv(NEDGES, T), T, 0, stream>>>(src, dst, dinv, norm, NEDGES);

    // ---- weight conversions ----
    k_f32_to_bf16<<<cdiv((long long)FEAT * HID, T), T, 0, stream>>>(W1, W1b, FEAT * HID);
    k_f32_to_bf16<<<cdiv((long long)HID * HID, T), T, 0, stream>>>(W2, W2b, HID * HID);
    k_f32_to_bf16<<<cdiv((long long)HID * OUTF, T), T, 0, stream>>>(W3, W3b, HID * OUTF);

    dim3 blk(T);

    // ---- layer 1: h = nodes @ W1 ; agg ; relu -> xb ----
    k_f32_to_bf16<<<cdiv((long long)NNODES * FEAT, T), T, 0, stream>>>(nodes, xb, NNODES * FEAT);
    {
        dim3 grid(cdiv(NNODES, BM), cdiv(HID, BN));
        k_gemm_bf16<<<grid, blk, 0, stream>>>(xb, W1b, h, NNODES, HID, FEAT);
    }
    k_agg_selfinit<<<cdiv((long long)NNODES * HID, T), T, 0, stream>>>(h, dinv, agg, NNODES, HID);
    k_agg_edges<<<NEDGES, T, 0, stream>>>(h, src, dst, norm, agg, HID);
    k_bias_relu_bf16<<<cdiv((long long)NNODES * HID, T), T, 0, stream>>>(agg, b1, xb, NNODES, HID);

    // ---- layer 2: h = x1 @ W2 ; agg ; relu -> xb ----
    {
        dim3 grid(cdiv(NNODES, BM), cdiv(HID, BN));
        k_gemm_bf16<<<grid, blk, 0, stream>>>(xb, W2b, h, NNODES, HID, HID);
    }
    k_agg_selfinit<<<cdiv((long long)NNODES * HID, T), T, 0, stream>>>(h, dinv, agg, NNODES, HID);
    k_agg_edges<<<NEDGES, T, 0, stream>>>(h, src, dst, norm, agg, HID);
    k_bias_relu_bf16<<<cdiv((long long)NNODES * HID, T), T, 0, stream>>>(agg, b2, xb, NNODES, HID);

    // ---- layer 3: emb = x2 @ W3 ; agg ; +b3 (no relu) ----
    {
        dim3 grid(cdiv(NNODES, BM), cdiv(OUTF, BN));
        k_gemm_bf16<<<grid, blk, 0, stream>>>(xb, W3b, h, NNODES, OUTF, HID);
    }
    k_agg_selfinit<<<cdiv((long long)NNODES * OUTF, T), T, 0, stream>>>(h, dinv, agg, NNODES, OUTF);
    k_agg_edges<<<NEDGES, T, 0, stream>>>(h, src, dst, norm, agg, OUTF);
    k_bias_f32<<<cdiv((long long)NNODES * OUTF, T), T, 0, stream>>>(agg, b3, NNODES, OUTF);

    // ---- normalize + similarity GEMM ----
    k_l2norm_pairs_T<<<NPAIRS, T, 0, stream>>>(agg, embT);
    k_l2norm_img<<<BATCH, T, 0, stream>>>(img, imgb);
    {
        dim3 grid(cdiv(BATCH, BM), cdiv(NPAIRS, BN));
        k_gemm_bf16<<<grid, blk, 0, stream>>>(imgb, embT, out, BATCH, NPAIRS, OUTF);
    }
}
